// Attention_84043920048344
// MI455X (gfx1250) — matrix operand order repaired and checked by reading, not compiled, not run on hardware
//
#include <hip/hip_runtime.h>
#include <hip/hip_bf16.h>

typedef _Float16 v16h __attribute__((ext_vector_type(16)));
typedef _Float16 v8h  __attribute__((ext_vector_type(8)));
typedef float    v8f  __attribute__((ext_vector_type(8)));
typedef float    f2v  __attribute__((ext_vector_type(2)));

#define H_DIM 512
#define L_DIM 2048
#define N_DIM 128
#define LT    64
#define CLIP  10.0f

__device__ __forceinline__ float tanh_fast(float x) {
#if __has_builtin(__builtin_amdgcn_tanhf)
    return __builtin_amdgcn_tanhf(x);
#else
    float ax = fabsf(x);
    float e  = __expf(2.0f * ax);
    float t  = 1.0f - 2.0f / (e + 1.0f);
    return copysignf(t, x);
#endif
}

// ---------------------------------------------------------------------------
// Kernel 1: q = query @ Wq^T + bq   (N x H, tiny: 67 MFLOP, scalar is fine)
// ---------------------------------------------------------------------------
__global__ __launch_bounds__(256) void qproj_kernel(
    const float* __restrict__ query, const float* __restrict__ Wq,
    const float* __restrict__ bq, float* __restrict__ qout)
{
    const int id = blockIdx.x * 256 + threadIdx.x;   // 0 .. N*H-1
    const int n  = id >> 9;            // H_DIM == 512
    const int o  = id & (H_DIM - 1);
    const float4* a = (const float4*)(query + (size_t)n * H_DIM);
    const float4* w = (const float4*)(Wq    + (size_t)o * H_DIM);
    float s = 0.0f;
#pragma unroll 4
    for (int i = 0; i < H_DIM / 4; ++i) {
        float4 x = a[i], y = w[i];
        s += x.x * y.x + x.y * y.y + x.z * y.z + x.w * y.w;
    }
    qout[id] = s + bq[o];
}

// ---------------------------------------------------------------------------
// Kernel 2: split Wr (f32) into hi/lo f16 planes (once; L2-resident after)
// ---------------------------------------------------------------------------
__global__ __launch_bounds__(256) void split_kernel(
    const float* __restrict__ Wr, _Float16* __restrict__ hi, _Float16* __restrict__ lo)
{
    const int id = blockIdx.x * 256 + threadIdx.x;   // H*H threads exactly
    const float w = Wr[id];
    const _Float16 h = (_Float16)w;
    hi[id] = h;
    lo[id] = (_Float16)(w - (float)h);
}

// ---------------------------------------------------------------------------
// Kernel 3: fused  r[n] = Wr @ ref[n]^T + br  and  logits[n,l]
// grid (L/LT, N), 1024 threads = 32 waves; wave w owns output rows 16w..16w+15
// and 4 col-tiles of 16; K loop over H in steps of 32 (WMMA f16 K).
// ---------------------------------------------------------------------------
__global__ __launch_bounds__(1024) void fused_gemm_logits(
    const float* __restrict__ ref,
    const _Float16* __restrict__ wrhi,
    const _Float16* __restrict__ wrlo,
    const float* __restrict__ br,
    const float* __restrict__ vvec,
    const float* __restrict__ qmat,
    float* __restrict__ out)
{
    __shared__ alignas(64) _Float16 sBhi[LT][48];  // ref tile, hi halves (padded rows)
    __shared__ alignas(64) _Float16 sBlo[LT][48];  // ref tile, lo halves
    __shared__ float sLog[LT];                     // logit partials per column

    const int n    = blockIdx.y;
    const int l0   = blockIdx.x * LT;
    const int tid  = threadIdx.x;
    const int lane = tid & 31;
    const int wave = tid >> 5;

    if (tid < LT) sLog[tid] = 0.0f;

    const int mrow = lane & 15;        // M row / N col within a 16x16 tile
    const int half = lane >> 4;        // lane half selects K sub-range
    const int ka0  = half * 16;        // A frag: K = ka0..ka0+15 contiguous
    const int kb0  = half * 16;        // B frag: K = kb0..kb0+15 contiguous
    const int orow = wave * 16 + mrow; // Wr row this lane feeds into A

    // cooperative LDS fill coordinates: 1024 threads x float2 = 64 x 32 tile
    const int fl = tid >> 4;
    const int fk = (tid & 15) * 2;
    const float* srcRow = ref + ((size_t)n * L_DIM + (l0 + fl)) * H_DIM + fk;

    v8f acc[4];
#pragma unroll
    for (int t = 0; t < 4; ++t) acc[t] = (v8f){0.f,0.f,0.f,0.f,0.f,0.f,0.f,0.f};

    for (int kb = 0; kb < H_DIM; kb += 32) {
        __syncthreads();                                  // prev iter reads done
        // ---- stage ref tile, converting f32 -> (hi,lo) f16 exactly once ----
        f2v d = __builtin_nontemporal_load((const f2v*)(srcRow + kb));
        _Float16 h0 = (_Float16)d.x;
        _Float16 h1 = (_Float16)d.y;
        sBhi[fl][fk]     = h0;
        sBhi[fl][fk + 1] = h1;
        sBlo[fl][fk]     = (_Float16)(d.x - (float)h0);
        sBlo[fl][fk + 1] = (_Float16)(d.y - (float)h1);
        __syncthreads();

        // ---- A fragments from pre-split Wr (L2-resident) ----
        const _Float16* ah = wrhi + (size_t)orow * H_DIM + kb + ka0;
        const _Float16* al = wrlo + (size_t)orow * H_DIM + kb + ka0;
        v8h a0 = *(const v8h*)(ah);
        v8h a1 = *(const v8h*)(ah + 8);
        v8h c0 = *(const v8h*)(al);
        v8h c1 = *(const v8h*)(al + 8);
        v16h Ahi = __builtin_shufflevector(a0, a1, 0,1,2,3,4,5,6,7,8,9,10,11,12,13,14,15);
        v16h Alo = __builtin_shufflevector(c0, c1, 0,1,2,3,4,5,6,7,8,9,10,11,12,13,14,15);

#pragma unroll
        for (int t = 0; t < 4; ++t) {
            const int lcol = t * 16 + mrow;
            v16h Bh = *(const v16h*)&sBhi[lcol][kb0];
            v16h Bl = *(const v16h*)&sBlo[lcol][kb0];
            // split-f16 x3: hi*hi + hi*lo + lo*hi  (~f32 fidelity, f32 accum)
            acc[t] = __builtin_amdgcn_wmma_f32_16x16x32_f16(false, Ahi, false, Bh, (short)0, acc[t], false, false);
            acc[t] = __builtin_amdgcn_wmma_f32_16x16x32_f16(false, Ahi, false, Bl, (short)0, acc[t], false, false);
            acc[t] = __builtin_amdgcn_wmma_f32_16x16x32_f16(false, Alo, false, Bh, (short)0, acc[t], false, false);
        }
    }

    // ---- epilogue: r = acc + br (streamed NT store) and logit partials ----
    const int mofs = half * 8;         // C/D layout: lanes>=16 hold M = r+8
    float brv[8], vv8[8], qv8[8];
#pragma unroll
    for (int r = 0; r < 8; ++r) {
        const int o = wave * 16 + mofs + r;
        brv[r] = br[o];
        vv8[r] = vvec[o];
        qv8[r] = qmat[(size_t)n * H_DIM + o];
    }
#pragma unroll
    for (int t = 0; t < 4; ++t) {
        const int lcol = t * 16 + mrow;
        float* op = out + ((size_t)n * H_DIM + (wave * 16 + mofs)) * L_DIM + l0 + lcol;
        float pt = 0.0f;
#pragma unroll
        for (int r = 0; r < 8; ++r) {
            const float val = acc[t][r] + brv[r];
            __builtin_nontemporal_store(val, op + (size_t)r * L_DIM);
            pt += vv8[r] * tanh_fast(qv8[r] + val);
        }
        atomicAdd(&sLog[lcol], pt);    // ds_add_f32, 64 distinct addresses
    }
    __syncthreads();
    if (tid < LT) {
        const size_t logitsBase = (size_t)N_DIM * H_DIM * L_DIM;
        out[logitsBase + (size_t)n * L_DIM + l0 + tid] = CLIP * tanh_fast(sLog[tid]);
    }
}

// ---------------------------------------------------------------------------
extern "C" void kernel_launch(void* const* d_in, const int* in_sizes, int n_in,
                              void* d_out, int out_size, void* d_ws, size_t ws_size,
                              hipStream_t stream) {
    (void)in_sizes; (void)n_in; (void)out_size; (void)ws_size;
    const float* query = (const float*)d_in[0];
    const float* ref   = (const float*)d_in[1];
    const float* Wq    = (const float*)d_in[2];
    const float* bq    = (const float*)d_in[3];
    const float* Wr    = (const float*)d_in[4];
    const float* br    = (const float*)d_in[5];
    const float* v     = (const float*)d_in[6];
    float* out = (float*)d_out;

    // workspace: q (N*H f32) | Wr_hi (H*H f16) | Wr_lo (H*H f16)  ~1.3 MB
    float*    qbuf = (float*)d_ws;
    _Float16* wrhi = (_Float16*)((char*)d_ws + (size_t)N_DIM * H_DIM * sizeof(float));
    _Float16* wrlo = wrhi + (size_t)H_DIM * H_DIM;

    qproj_kernel<<<(N_DIM * H_DIM) / 256, 256, 0, stream>>>(query, Wq, bq, qbuf);
    split_kernel<<<(H_DIM * H_DIM) / 256, 256, 0, stream>>>(Wr, wrhi, wrlo);
    fused_gemm_logits<<<dim3(L_DIM / LT, N_DIM), 1024, 0, stream>>>(
        ref, wrhi, wrlo, br, v, qbuf, out);
}
